// Fine_Grain_Layer_19559281066237
// MI455X (gfx1250) — matrix operand order repaired and checked by reading, not compiled, not run on hardware
//
#include <hip/hip_runtime.h>
#include <hip/hip_bf16.h>
#include <math.h>

typedef _Float16 h16;
typedef __attribute__((ext_vector_type(16))) _Float16 v16h;
typedef __attribute__((ext_vector_type(8)))  _Float16 v8h;
typedef __attribute__((ext_vector_type(8)))  float    v8f;
typedef __attribute__((ext_vector_type(4)))  int      v4i;

#define DEV __device__ __forceinline__

static constexpr int N_NODES = 8192;
static constexpr int N_EDGES = 262144;

DEV v8f wmma32(v16h a, v16h b, v8f c) {
  return __builtin_amdgcn_wmma_f32_16x16x32_f16(false, a, false, b, (short)0, c,
                                                false, false);
}

// 16 contiguous halves (32B) -> v16h via two 16B vector loads
DEV v16h load16(const h16* p) {
  const v8h lo = *(const v8h*)(p);
  const v8h hi = *(const v8h*)(p + 8);
  return __builtin_shufflevector(lo, hi, 0, 1, 2, 3, 4, 5, 6, 7, 8, 9, 10, 11,
                                 12, 13, 14, 15);
}

// ---- CDNA5 async global->LDS copy (ASYNCcnt), guarded by builtin probe ----
#if __has_builtin(__builtin_amdgcn_global_load_async_to_lds_b128)
#define HAVE_ASYNC_LDS 1
typedef __attribute__((address_space(1))) v4i g_v4i;
typedef __attribute__((address_space(3))) v4i l_v4i;
DEV void async_copy16B(const h16* gsrc, h16* ldst) {
  __builtin_amdgcn_global_load_async_to_lds_b128((g_v4i*)gsrc, (l_v4i*)ldst, 0, 0);
}
DEV void async_wait() {
#if __has_builtin(__builtin_amdgcn_s_wait_asynccnt)
  __builtin_amdgcn_s_wait_asynccnt(0);
#else
  asm volatile("s_wait_asynccnt 0" ::: "memory");
#endif
}
#else
#define HAVE_ASYNC_LDS 0
DEV void async_copy16B(const h16* gsrc, h16* ldst) {
  *(v8h*)ldst = *(const v8h*)gsrc;
}
DEV void async_wait() {}
#endif

// A fragment element (row r, k) -> index into packed [ks][lane][16] layout.
// Contiguous k order used here: k%32 = 16*(lane/16) + element.
DEV int afrag_idx(int r, int k) {
  const int ks = k >> 5, kk = k & 31;
  const int g = (kk >> 4) & 1;
  const int i = kk & 15;
  return ks * 512 + ((r & 15) + (g << 4)) * 16 + i;
}

// 8-aligned K-chunk (row r, k multiple of 8) -> offset of 8 CONTIGUOUS halves
// in the packed A-fragment layout (see striping: each 8-chunk is contiguous).
DEV int afrag_chunk(int r, int k) {
  const int ks = k >> 5, kk = k & 31;
  const int g = (kk >> 4) & 1;
  const int hs = kk & 8;
  return ks * 512 + ((r & 15) + (g << 4)) * 16 + hs;
}

// A fragment: 16(M) x 32(K) gathered from row-major f16, row stride ld.
DEV v16h load_a_rm(const h16* p, int ld) {
  const int lane = threadIdx.x & 31;
  const int m = lane & 15, g = lane >> 4;
  v16h a;
#pragma unroll
  for (int i = 0; i < 16; ++i) {
    const int pr = i >> 1, w = i & 1;
    const int k = g * 16 + pr * 2 + w;
    a[i] = p[m * ld + k];
  }
  return a;
}

// LayerNorm one 64-wide row (+optional leaky-relu). Optional outputs:
// outF = f16 A-fragment-packed buffer (row `row`), outf = f32 row-major.
DEV void ln_row(const float* x, const float* gg, const float* bb, bool act,
                h16* outF, int row, float* outf) {
  float m = 0.f;
  for (int c = 0; c < 64; ++c) m += x[c];
  m *= (1.f / 64.f);
  float v = 0.f;
  for (int c = 0; c < 64; ++c) { float d = x[c] - m; v += d * d; }
  v *= (1.f / 64.f);
  const float rs = rsqrtf(v + 1e-5f);
  for (int c = 0; c < 64; ++c) {
    float y = (x[c] - m) * rs * gg[c] + bb[c];
    if (act) y = y > 0.f ? y : 0.01f * y;
    if (outF) outF[afrag_idx(row, c)] = (h16)y;
    if (outf) outf[c] = y;
  }
}

// ---------------- prep kernels ----------------
__global__ void k_f32_to_f16(const float* __restrict__ s, h16* __restrict__ d, int n) {
  int i = blockIdx.x * blockDim.x + threadIdx.x;
  if (i < n) d[i] = (h16)s[i];
}
__global__ void k_zero_f32(float* __restrict__ d, int n) {
  int i = blockIdx.x * blockDim.x + threadIdx.x;
  if (i < n) d[i] = 0.f;
}
// Pack W [Kreal x 64] (zero-padded to K) into B fragments:
// out[((ks*4+nt)*32 + lane)*16 + i] = W[(ks*32 + g*16 + i)*64 + nt*16 + n]
__global__ void k_pack_w(const float* __restrict__ W, h16* __restrict__ out,
                         int K, int Kreal) {
  int idx = blockIdx.x * blockDim.x + threadIdx.x;
  if (idx >= K * 64) return;
  const int i = idx & 15;
  const int lane = (idx >> 4) & 31;
  const int frag = idx >> 9;
  const int nt = frag & 3, ks = frag >> 2;
  const int n = lane & 15, g = lane >> 4;
  const int k = ks * 32 + g * 16 + i;
  out[idx] = (k < Kreal) ? (h16)W[k * 64 + nt * 16 + n] : (h16)0.f;
}

// ---------------- q/k/v projection: out = act? lrelu(h @ W) : h @ W --------
__global__ __launch_bounds__(32)
void k_qkv(const h16* __restrict__ hin, const h16* __restrict__ Wp,
           h16* __restrict__ out, int act) {
  const int lane = threadIdx.x;
  const int nl = lane & 15, g = lane >> 4;
  const int base = blockIdx.x * 16;
  v16h af[2];
#pragma unroll
  for (int ks = 0; ks < 2; ++ks)
    af[ks] = load_a_rm(&hin[(size_t)base * 64 + ks * 32], 64);
#pragma unroll
  for (int nt = 0; nt < 4; ++nt) {
    v8f acc = {};
#pragma unroll
    for (int ks = 0; ks < 2; ++ks)
      acc = wmma32(af[ks], load16(&Wp[((ks * 4 + nt) * 32 + lane) * 16]), acc);
#pragma unroll
    for (int j = 0; j < 8; ++j) {
      float y = acc[j];
      if (act) y = y > 0.f ? y : 0.01f * y;
      out[(size_t)(base + j + 8 * g) * 64 + nt * 16 + nl] = (h16)y;
    }
  }
}

// ---------------- edge MLP + coord coef + scatter ----------------
__global__ __launch_bounds__(32)
void k_edges(const float* __restrict__ coords, const h16* __restrict__ hh,
             const float* __restrict__ efeat, const int* __restrict__ src,
             const int* __restrict__ dst,
             const h16* __restrict__ eW1p, const float* __restrict__ eb1,
             const float* __restrict__ eg1, const float* __restrict__ ebt1,
             const h16* __restrict__ eW2p, const float* __restrict__ eb2,
             const float* __restrict__ eg2, const float* __restrict__ ebt2,
             const h16* __restrict__ cW1p, const float* __restrict__ cb1,
             const float* __restrict__ cg1, const float* __restrict__ cbt1,
             const float* __restrict__ cW2, const float* __restrict__ cb2,
             float* __restrict__ cnt, float* __restrict__ xup,
             float* __restrict__ aggr) {
  __shared__ __align__(32) h16 AinpF[6 * 512];   // 16x192 in A-fragment order
  __shared__ __align__(32) h16 A2F[2 * 512];     // 16x64 in A-fragment order
  __shared__ float Cbuf[16 * 64];
  __shared__ float Mbuf[16 * 64];
  __shared__ float xrel[16][3];
  __shared__ float coef[16];
  const int lane = threadIdx.x;
  const int nl = lane & 15, g = lane >> 4;
  const int e0 = blockIdx.x * 16;

  // h[src], h[dst]: 16 rows x 8 chunks-of-8; b128 both sides (or async DMA)
#pragma unroll
  for (int it = 0; it < 4; ++it) {
    const int cid = lane + it * 32;  // 0..127
    const int r = cid >> 3, ch = cid & 7;
    const int e = e0 + r;
    async_copy16B(&hh[(size_t)src[e] * 64 + ch * 8],
                  &AinpF[afrag_chunk(r, ch * 8)]);
    async_copy16B(&hh[(size_t)dst[e] * 64 + ch * 8],
                  &AinpF[afrag_chunk(r, 64 + ch * 8)]);
  }
  // efeat (f32 -> f16): 16 rows x 4 chunks-of-8
#pragma unroll
  for (int it = 0; it < 2; ++it) {
    const int cid = lane + it * 32;  // 0..63
    const int r = cid >> 2, ch = cid & 3;
    const float* pf = &efeat[(size_t)(e0 + r) * 32 + ch * 8];
    v8h d;
#pragma unroll
    for (int e2 = 0; e2 < 8; ++e2) d[e2] = (h16)pf[e2];
    *(v8h*)&AinpF[afrag_chunk(r, 128 + ch * 8)] = d;
  }
  if (lane < 16) {
    const int e = e0 + lane;
    const int s = src[e], d = dst[e];
    float d2 = 0.f;
#pragma unroll
    for (int ax = 0; ax < 3; ++ax) {
      const float r = coords[s * 3 + ax] - coords[d * 3 + ax];
      xrel[lane][ax] = r;
      d2 += r * r;
    }
    float sig = 1.0f;
#pragma unroll
    for (int i = 0; i < 15; ++i) {
      AinpF[afrag_idx(lane, 160 + i)] = (h16)__expf(-d2 / sig);
      sig *= 1.5f;
    }
#pragma unroll
    for (int i = 175; i < 192; ++i) AinpF[afrag_idx(lane, i)] = (h16)0.f;
  }
  async_wait();
  __syncthreads();

  // GEMM1: [16x192] @ [192x64]
  {
    v16h af[6];
#pragma unroll
    for (int ks = 0; ks < 6; ++ks) af[ks] = load16(&AinpF[ks * 512 + lane * 16]);
#pragma unroll
    for (int nt = 0; nt < 4; ++nt) {
      v8f acc = {};
#pragma unroll
      for (int ks = 0; ks < 6; ++ks)
        acc = wmma32(af[ks], load16(&eW1p[((ks * 4 + nt) * 32 + lane) * 16]), acc);
      const float bias = eb1[nt * 16 + nl];
#pragma unroll
      for (int j = 0; j < 8; ++j)
        Cbuf[(j + 8 * g) * 64 + nt * 16 + nl] = acc[j] + bias;
    }
  }
  __syncthreads();
  if (lane < 16) ln_row(&Cbuf[lane * 64], eg1, ebt1, true, A2F, lane, nullptr);
  __syncthreads();

  // GEMM2: msg = LN(lrelu(.) @ eW2 + eb2)
  {
    v16h af[2];
#pragma unroll
    for (int ks = 0; ks < 2; ++ks) af[ks] = load16(&A2F[ks * 512 + lane * 16]);
#pragma unroll
    for (int nt = 0; nt < 4; ++nt) {
      v8f acc = {};
#pragma unroll
      for (int ks = 0; ks < 2; ++ks)
        acc = wmma32(af[ks], load16(&eW2p[((ks * 4 + nt) * 32 + lane) * 16]), acc);
      const float bias = eb2[nt * 16 + nl];
#pragma unroll
      for (int j = 0; j < 8; ++j)
        Cbuf[(j + 8 * g) * 64 + nt * 16 + nl] = acc[j] + bias;
    }
  }
  __syncthreads();
  if (lane < 16)
    ln_row(&Cbuf[lane * 64], eg2, ebt2, false, A2F, lane, &Mbuf[lane * 64]);
  __syncthreads();

  // GEMM3: coef path
  {
    v16h af[2];
#pragma unroll
    for (int ks = 0; ks < 2; ++ks) af[ks] = load16(&A2F[ks * 512 + lane * 16]);
#pragma unroll
    for (int nt = 0; nt < 4; ++nt) {
      v8f acc = {};
#pragma unroll
      for (int ks = 0; ks < 2; ++ks)
        acc = wmma32(af[ks], load16(&cW1p[((ks * 4 + nt) * 32 + lane) * 16]), acc);
      const float bias = cb1[nt * 16 + nl];
#pragma unroll
      for (int j = 0; j < 8; ++j)
        Cbuf[(j + 8 * g) * 64 + nt * 16 + nl] = acc[j] + bias;
    }
  }
  __syncthreads();
  if (lane < 16) {
    ln_row(&Cbuf[lane * 64], cg1, cbt1, true, nullptr, 0, &Cbuf[lane * 64]);
    float acc = cb2[0];
    for (int c = 0; c < 64; ++c) acc += Cbuf[lane * 64 + c] * cW2[c];
    coef[lane] = acc;
  }
  __syncthreads();

  // scatter (segment sums)
  if (lane < 16) {
    const int e = e0 + lane;
    const int d = dst[e];
    atomicAdd(&cnt[d], 1.0f);
#pragma unroll
    for (int ax = 0; ax < 3; ++ax)
      atomicAdd(&xup[d * 3 + ax], xrel[lane][ax] * coef[lane]);
  }
  for (int t = lane; t < 16 * 64; t += 32) {
    const int r = t >> 6, c = t & 63;
    atomicAdd(&aggr[(size_t)dst[e0 + r] * 64 + c], Mbuf[r * 64 + c]);
  }
}

// ---------------- flash attention: attA = softmax(mask*QK^T - 1000(1-mask)) @ V
__global__ __launch_bounds__(32)
void k_attn(const h16* __restrict__ qA, const h16* __restrict__ kB,
            const h16* __restrict__ vB, const float* __restrict__ mask,
            float* __restrict__ attA) {
  __shared__ __align__(32) h16 Kt[32 * 64];   // [key][feat] row-major
  __shared__ __align__(32) h16 VtT[64 * 32];  // [feat][key] transposed
  __shared__ __align__(32) h16 PtF[512];      // 16x32 P in A-fragment order
  __shared__ float St[16 * 32];
  __shared__ float mrow[16], lrow[16], srow[16];
  const int lane = threadIdx.x;
  const int nl = lane & 15, g = lane >> 4;
  const int qb = blockIdx.x * 16;

  const v16h a0 = load_a_rm(&qA[(size_t)qb * 64], 64);
  const v16h a1 = load_a_rm(&qA[(size_t)qb * 64 + 32], 64);
  v8f O[4] = {};
  if (lane < 16) { mrow[lane] = -1e30f; lrow[lane] = 0.f; }
  __syncthreads();

  for (int kt = 0; kt < 256; ++kt) {
    const int kb = kt * 32;
    {  // lane owns key row `lane`
      const h16* krow = &kB[(size_t)(kb + lane) * 64];
#pragma unroll
      for (int fb = 0; fb < 8; ++fb)
        async_copy16B(krow + fb * 8, &Kt[lane * 64 + fb * 8]);
      const v8h* vrow = (const v8h*)&vB[(size_t)(kb + lane) * 64];
#pragma unroll
      for (int fb = 0; fb < 8; ++fb) {
        const v8h d = vrow[fb];
#pragma unroll
        for (int e = 0; e < 8; ++e) VtT[(fb * 8 + e) * 32 + lane] = d[e];
      }
    }
    if (kt + 1 < 256) {  // prefetch next mask tile (dominant HBM stream)
      __builtin_prefetch(&mask[(size_t)(qb + nl) * N_NODES + kb + 32 + g * 16],
                         0, 1);
      __builtin_prefetch(&kB[(size_t)(kb + 32 + lane) * 64], 0, 1);
      __builtin_prefetch(&vB[(size_t)(kb + 32 + lane) * 64], 0, 1);
    }
    async_wait();
    __syncthreads();

    // S = Q @ K^T, masked
#pragma unroll
    for (int nt = 0; nt < 2; ++nt) {
      v8f s = {};
      s = wmma32(a0, load16(&Kt[(nt * 16 + nl) * 64 + g * 16]), s);
      s = wmma32(a1, load16(&Kt[(nt * 16 + nl) * 64 + 32 + g * 16]), s);
#pragma unroll
      for (int j = 0; j < 8; ++j) {
        const int row = qb + j + 8 * g;
        const int col = kb + nt * 16 + nl;
        const float mv = mask[(size_t)row * N_NODES + col];
        St[(j + 8 * g) * 32 + nt * 16 + nl] = mv * s[j] - 1000.f * (1.f - mv);
      }
    }
    __syncthreads();

    // online softmax per row; P written in A-fragment order
    if (lane < 16) {
      const float mo = mrow[lane];
      float mn = mo;
      for (int c = 0; c < 32; ++c) mn = fmaxf(mn, St[lane * 32 + c]);
      const float sc = __expf(mo - mn);
      float l = lrow[lane] * sc;
      for (int c = 0; c < 32; ++c) {
        const float p = __expf(St[lane * 32 + c] - mn);
        l += p;
        PtF[afrag_idx(lane, c)] = (h16)p;
      }
      mrow[lane] = mn; lrow[lane] = l; srow[lane] = sc;
    }
    __syncthreads();

    // O = O*scale + P @ V
    const v16h ap = load16(&PtF[lane * 16]);
#pragma unroll
    for (int nt = 0; nt < 4; ++nt) {
      v8f o = O[nt];
#pragma unroll
      for (int j = 0; j < 8; ++j) o[j] *= srow[j + 8 * g];
      O[nt] = wmma32(ap, load16(&VtT[(nt * 16 + nl) * 32 + g * 16]), o);
    }
    __syncthreads();
  }

#pragma unroll
  for (int nt = 0; nt < 4; ++nt)
#pragma unroll
    for (int j = 0; j < 8; ++j)
      attA[(size_t)(qb + j + 8 * g) * 64 + nt * 16 + nl] =
          O[nt][j] / lrow[j + 8 * g];
}

// ---------------- node update (+ x_new) ----------------
__global__ __launch_bounds__(32)
void k_node(const float* __restrict__ coords, const float* __restrict__ orig_coords,
            const float* __restrict__ hfeat, const h16* __restrict__ hh,
            const float* __restrict__ orig_feats, const float* __restrict__ attA,
            int use_att, const float* __restrict__ cnt,
            const float* __restrict__ xup, const float* __restrict__ aggr,
            const h16* __restrict__ nW1p, const float* __restrict__ nb1,
            const float* __restrict__ ng1, const float* __restrict__ nbt1,
            const h16* __restrict__ nW2p, const float* __restrict__ nb2,
            const float* __restrict__ ng2, const float* __restrict__ nbt2,
            float* __restrict__ out_x, float* __restrict__ out_h) {
  __shared__ __align__(32) h16 AtiF[8 * 512];  // 16x256 in A-fragment order
  __shared__ __align__(32) h16 A2F[2 * 512];
  __shared__ float Cbuf[16 * 64];
  const int lane = threadIdx.x;
  const int nl = lane & 15, g = lane >> 4;
  const int base = blockIdx.x * 16;

  if (lane < 16) {
    const int nd = base + lane;
    const float cv = fmaxf(cnt[nd], 1.0f);
#pragma unroll
    for (int ax = 0; ax < 3; ++ax)
      out_x[nd * 3 + ax] = 0.25f * orig_coords[nd * 3 + ax] +
                           0.75f * coords[nd * 3 + ax] + xup[nd * 3 + ax] / cv;
  }
  // stage 16x256 input: chunks-of-8, vectorized
#pragma unroll
  for (int it = 0; it < 4; ++it) {
    const int cid = lane + it * 32;  // 0..127
    const int r = cid >> 3, ch = cid & 7;
    const int nd = base + r;
    async_copy16B(&hh[(size_t)nd * 64 + ch * 8], &AtiF[afrag_chunk(r, ch * 8)]);
    const float inv = 1.0f / fmaxf(cnt[nd], 1.0f);
    const float* pa = &aggr[(size_t)nd * 64 + ch * 8];
    const float* po = &orig_feats[(size_t)nd * 64 + ch * 8];
    v8h da, dt, do_;
    if (use_att) {
      const float* pt = &attA[(size_t)nd * 64 + ch * 8];
#pragma unroll
      for (int e2 = 0; e2 < 8; ++e2) dt[e2] = (h16)pt[e2];
    } else {
#pragma unroll
      for (int e2 = 0; e2 < 8; ++e2) dt[e2] = (h16)0.f;
    }
#pragma unroll
    for (int e2 = 0; e2 < 8; ++e2) {
      da[e2]  = (h16)(pa[e2] * inv);
      do_[e2] = (h16)po[e2];
    }
    *(v8h*)&AtiF[afrag_chunk(r, 64 + ch * 8)]  = da;
    *(v8h*)&AtiF[afrag_chunk(r, 128 + ch * 8)] = dt;
    *(v8h*)&AtiF[afrag_chunk(r, 192 + ch * 8)] = do_;
  }
  async_wait();
  __syncthreads();

  {
    v16h af[8];
#pragma unroll
    for (int ks = 0; ks < 8; ++ks) af[ks] = load16(&AtiF[ks * 512 + lane * 16]);
#pragma unroll
    for (int nt = 0; nt < 4; ++nt) {
      v8f acc = {};
#pragma unroll
      for (int ks = 0; ks < 8; ++ks)
        acc = wmma32(af[ks], load16(&nW1p[((ks * 4 + nt) * 32 + lane) * 16]), acc);
      const float bias = nb1[nt * 16 + nl];
#pragma unroll
      for (int j = 0; j < 8; ++j)
        Cbuf[(j + 8 * g) * 64 + nt * 16 + nl] = acc[j] + bias;
    }
  }
  __syncthreads();
  if (lane < 16) ln_row(&Cbuf[lane * 64], ng1, nbt1, true, A2F, lane, nullptr);
  __syncthreads();

  {
    v16h af[2];
#pragma unroll
    for (int ks = 0; ks < 2; ++ks) af[ks] = load16(&A2F[ks * 512 + lane * 16]);
#pragma unroll
    for (int nt = 0; nt < 4; ++nt) {
      v8f acc = {};
#pragma unroll
      for (int ks = 0; ks < 2; ++ks)
        acc = wmma32(af[ks], load16(&nW2p[((ks * 4 + nt) * 32 + lane) * 16]), acc);
      const float bias = nb2[nt * 16 + nl];
#pragma unroll
      for (int j = 0; j < 8; ++j)
        Cbuf[(j + 8 * g) * 64 + nt * 16 + nl] = acc[j] + bias;
    }
  }
  __syncthreads();
  if (lane < 16) {
    ln_row(&Cbuf[lane * 64], ng2, nbt2, false, nullptr, 0, &Cbuf[lane * 64]);
    const int nd = base + lane;
    for (int c = 0; c < 64; ++c)
      out_h[(size_t)nd * 64 + c] =
          0.5f * Cbuf[lane * 64 + c] + 0.5f * hfeat[(size_t)nd * 64 + c];
  }
}

// ---------------- host ----------------
extern "C" void kernel_launch(void* const* d_in, const int* in_sizes, int n_in,
                              void* d_out, int out_size, void* d_ws, size_t ws_size,
                              hipStream_t stream) {
  const float* coords_A = (const float*)d_in[0];
  const float* h_A      = (const float*)d_in[1];
  const float* origf_A  = (const float*)d_in[2];
  const float* origc_A  = (const float*)d_in[3];
  const float* ef_A     = (const float*)d_in[4];
  const float* coords_B = (const float*)d_in[5];
  const float* h_B      = (const float*)d_in[6];
  const float* origf_B  = (const float*)d_in[7];
  const float* origc_B  = (const float*)d_in[8];
  const float* ef_B     = (const float*)d_in[9];
  const float* mask     = (const float*)d_in[10];
  const float* eW1  = (const float*)d_in[11];
  const float* eb1  = (const float*)d_in[12];
  const float* eg1  = (const float*)d_in[13];
  const float* ebt1 = (const float*)d_in[14];
  const float* eW2  = (const float*)d_in[15];
  const float* eb2  = (const float*)d_in[16];
  const float* eg2  = (const float*)d_in[17];
  const float* ebt2 = (const float*)d_in[18];
  const float* qW   = (const float*)d_in[19];
  const float* kW   = (const float*)d_in[20];
  const float* vW   = (const float*)d_in[21];
  const float* nW1  = (const float*)d_in[22];
  const float* nb1  = (const float*)d_in[23];
  const float* ng1  = (const float*)d_in[24];
  const float* nbt1 = (const float*)d_in[25];
  const float* nW2  = (const float*)d_in[26];
  const float* nb2  = (const float*)d_in[27];
  const float* ng2  = (const float*)d_in[28];
  const float* nbt2 = (const float*)d_in[29];
  const float* cW1  = (const float*)d_in[30];
  const float* cb1  = (const float*)d_in[31];
  const float* cg1  = (const float*)d_in[32];
  const float* cbt1 = (const float*)d_in[33];
  const float* cW2  = (const float*)d_in[34];
  const float* cb2  = (const float*)d_in[35];
  const int* src_A  = (const int*)d_in[36];
  const int* dst_A  = (const int*)d_in[37];
  const int* src_B  = (const int*)d_in[38];
  const int* dst_B  = (const int*)d_in[39];

  float* outxA = (float*)d_out;
  float* outhA = outxA + (size_t)N_NODES * 3;
  float* outxB = outhA + (size_t)N_NODES * 64;
  float* outhB = outxB + (size_t)N_NODES * 3;

  size_t off = 0;
  auto carve = [&](size_t bytes) -> char* {
    char* p = (char*)d_ws + off;
    off = (off + bytes + 255) & ~(size_t)255;
    return p;
  };
  h16* eW1p = (h16*)carve(192 * 64 * 2);
  h16* eW2p = (h16*)carve(64 * 64 * 2);
  h16* cW1p = (h16*)carve(64 * 64 * 2);
  h16* nW1p = (h16*)carve(256 * 64 * 2);
  h16* nW2p = (h16*)carve(64 * 64 * 2);
  h16* qWp  = (h16*)carve(64 * 64 * 2);
  h16* kWp  = (h16*)carve(64 * 64 * 2);
  h16* vWp  = (h16*)carve(64 * 64 * 2);
  h16* hAh  = (h16*)carve((size_t)N_NODES * 64 * 2);
  h16* hBh  = (h16*)carve((size_t)N_NODES * 64 * 2);
  h16* qAh  = (h16*)carve((size_t)N_NODES * 64 * 2);
  h16* kBh  = (h16*)carve((size_t)N_NODES * 64 * 2);
  h16* vBh  = (h16*)carve((size_t)N_NODES * 64 * 2);
  float* attA  = (float*)carve((size_t)N_NODES * 64 * 4);
  float* cntA  = (float*)carve((size_t)N_NODES * 4);
  float* cntB  = (float*)carve((size_t)N_NODES * 4);
  float* xupA  = (float*)carve((size_t)N_NODES * 3 * 4);
  float* xupB  = (float*)carve((size_t)N_NODES * 3 * 4);
  float* aggrA = (float*)carve((size_t)N_NODES * 64 * 4);
  float* aggrB = (float*)carve((size_t)N_NODES * 64 * 4);
  (void)ws_size; (void)in_sizes; (void)n_in; (void)out_size;

  auto packw = [&](const float* s, h16* d, int K, int Kreal) {
    k_pack_w<<<(K * 64 + 255) / 256, 256, 0, stream>>>(s, d, K, Kreal);
  };
  auto cvt = [&](const float* s, h16* d, int n) {
    k_f32_to_f16<<<(n + 255) / 256, 256, 0, stream>>>(s, d, n);
  };
  auto zero = [&](float* d, int n) {
    k_zero_f32<<<(n + 255) / 256, 256, 0, stream>>>(d, n);
  };

  packw(eW1, eW1p, 192, 175);
  packw(eW2, eW2p, 64, 64);
  packw(cW1, cW1p, 64, 64);
  packw(nW1, nW1p, 256, 256);
  packw(nW2, nW2p, 64, 64);
  packw(qW, qWp, 64, 64);
  packw(kW, kWp, 64, 64);
  packw(vW, vWp, 64, 64);
  cvt(h_A, hAh, N_NODES * 64);
  cvt(h_B, hBh, N_NODES * 64);
  zero(cntA, N_NODES);  zero(cntB, N_NODES);
  zero(xupA, N_NODES * 3); zero(xupB, N_NODES * 3);
  zero(aggrA, N_NODES * 64); zero(aggrB, N_NODES * 64);

  k_qkv<<<N_NODES / 16, 32, 0, stream>>>(hAh, qWp, qAh, 1);
  k_qkv<<<N_NODES / 16, 32, 0, stream>>>(hBh, kWp, kBh, 1);
  k_qkv<<<N_NODES / 16, 32, 0, stream>>>(hBh, vWp, vBh, 0);

  k_edges<<<N_EDGES / 16, 32, 0, stream>>>(
      coords_A, hAh, ef_A, src_A, dst_A, eW1p, eb1, eg1, ebt1, eW2p, eb2, eg2,
      ebt2, cW1p, cb1, cg1, cbt1, cW2, cb2, cntA, xupA, aggrA);
  k_edges<<<N_EDGES / 16, 32, 0, stream>>>(
      coords_B, hBh, ef_B, src_B, dst_B, eW1p, eb1, eg1, ebt1, eW2p, eb2, eg2,
      ebt2, cW1p, cb1, cg1, cbt1, cW2, cb2, cntB, xupB, aggrB);

  k_attn<<<N_NODES / 16, 32, 0, stream>>>(qAh, kBh, vBh, mask, attA);

  k_node<<<N_NODES / 16, 32, 0, stream>>>(
      coords_A, origc_A, h_A, hAh, origf_A, attA, 1, cntA, xupA, aggrA, nW1p,
      nb1, ng1, nbt1, nW2p, nb2, ng2, nbt2, outxA, outhA);
  k_node<<<N_NODES / 16, 32, 0, stream>>>(
      coords_B, origc_B, h_B, hBh, origf_B, attA, 0, cntB, xupB, aggrB, nW1p,
      nb1, ng1, nbt1, nW2p, nb2, ng2, nbt2, outxB, outhB);
}
